// FasterRcnnPredictModel_54881092108513
// MI455X (gfx1250) — compile-verified, matrix operand order checked by reading, not run on hardware
//
#include <hip/hip_runtime.h>
#include <float.h>

#define NN 2000
#define NUM_CLASSES 21
#define MAX_PER_CLASS 50
#define MAX_PER_IMAGE 100
#define NEGV (-1000000000.0f)
#define NTILES (NN / 16)       // 125
#define RS_B 2080              // LDS row stride in bytes (2000 padded)
#define RS_D (RS_B / 4)        // LDS row stride in dwords

#ifndef USE_ASYNC_LDS
#define USE_ASYNC_LDS 1        // gfx1250 global_load_async_to_lds_b32 path
#endif

typedef __attribute__((ext_vector_type(8))) int v8i;

// ---------------------------------------------------------------- decode ----
__global__ void decode_kernel(const float* __restrict__ props,
                              const float* __restrict__ deltas,
                              float4* __restrict__ boxes) {
  int i = blockIdx.x * blockDim.x + threadIdx.x;
  if (i >= NN) return;
  float p0 = props[i * 4 + 0], p1 = props[i * 4 + 1];
  float p2 = props[i * 4 + 2], p3 = props[i * 4 + 3];
  float h = p2 - p0, w = p3 - p1;
  float cy = p0 + 0.5f * h, cx = p1 + 0.5f * w;
  float dy = deltas[i * 4 + 0], dx = deltas[i * 4 + 1];
  float dh = deltas[i * 4 + 2], dw = deltas[i * 4 + 3];
  float pcy = dy * h + cy, pcx = dx * w + cx;
  float ph = h * expf(dh), pw = w * expf(dw);
  boxes[i] = make_float4(pcy - 0.5f * ph, pcx - 0.5f * pw,
                         pcy + 0.5f * ph, pcx + 0.5f * pw);
}

// ------------------------------------------------------- IoU > thr mask ----
__global__ void iou_mask_kernel(const float4* __restrict__ boxes,
                                unsigned char* __restrict__ maskM) {
  long long idx = (long long)blockIdx.x * blockDim.x + threadIdx.x;
  if (idx >= (long long)NN * NN) return;
  int i = (int)(idx / NN), j = (int)(idx % NN);
  float4 bi = boxes[i], bj = boxes[j];
  float ai = (bi.z - bi.x) * (bi.w - bi.y);
  float aj = (bj.z - bj.x) * (bj.w - bj.y);
  float iy1 = fmaxf(bi.x, bj.x), ix1 = fmaxf(bi.y, bj.y);
  float iy2 = fminf(bi.z, bj.z), ix2 = fminf(bi.w, bj.w);
  float inter = fmaxf(iy2 - iy1, 0.0f) * fmaxf(ix2 - ix1, 0.0f);
  float uni = fmaxf(ai + aj - inter, 1e-8f);
  maskM[idx] = (inter / uni) > 0.7f ? (unsigned char)1 : (unsigned char)0;
}

// ------------------------------------------- per-class bitonic argsort ----
__global__ __launch_bounds__(1024)
void sort_kernel(const float* __restrict__ score,     // [NN, NUM_CLASSES]
                 int* __restrict__ orderAll,          // [C, NN]
                 float* __restrict__ scoreSorted) {   // [C, NN]
  __shared__ float k_s[2048];
  __shared__ int i_s[2048];
  int c = blockIdx.x;
  int t = threadIdx.x;
  for (int p = t; p < 2048; p += 1024) {
    if (p < NN) { k_s[p] = score[p * NUM_CLASSES + c]; }
    else        { k_s[p] = -FLT_MAX; }
    i_s[p] = p;
  }
  __syncthreads();
  for (int k = 2; k <= 2048; k <<= 1) {
    for (int j = k >> 1; j > 0; j >>= 1) {
      int i = ((t & ~(j - 1)) << 1) | (t & (j - 1));
      int ixj = i | j;
      bool up = ((i & k) == 0);
      float s0 = k_s[i], s1 = k_s[ixj];
      int d0 = i_s[i], d1 = i_s[ixj];
      bool before01 = (s0 > s1) || (s0 == s1 && d0 < d1);
      bool doswap = up ? !before01 : before01;
      if (doswap) {
        k_s[i] = s1; k_s[ixj] = s0;
        i_s[i] = d1; i_s[ixj] = d0;
      }
      __syncthreads();
    }
  }
  for (int p = t; p < NN; p += 1024) {
    orderAll[c * NN + p] = i_s[p];
    scoreSorted[c * NN + p] = k_s[p];
  }
}

// packs 16 bytes of one staged-row column into 4 dwords
__device__ __forceinline__ void packCol(const unsigned char* col,
                                        unsigned int& w0, unsigned int& w1,
                                        unsigned int& w2, unsigned int& w3) {
  w0 = (unsigned int)col[0 * RS_B]
     | ((unsigned int)col[1 * RS_B] << 8)
     | ((unsigned int)col[2 * RS_B] << 16)
     | ((unsigned int)col[3 * RS_B] << 24);
  w1 = (unsigned int)col[4 * RS_B]
     | ((unsigned int)col[5 * RS_B] << 8)
     | ((unsigned int)col[6 * RS_B] << 16)
     | ((unsigned int)col[7 * RS_B] << 24);
  w2 = (unsigned int)col[8 * RS_B]
     | ((unsigned int)col[9 * RS_B] << 8)
     | ((unsigned int)col[10 * RS_B] << 16)
     | ((unsigned int)col[11 * RS_B] << 24);
  w3 = (unsigned int)col[12 * RS_B]
     | ((unsigned int)col[13 * RS_B] << 8)
     | ((unsigned int)col[14 * RS_B] << 16)
     | ((unsigned int)col[15 * RS_B] << 24);
}

// -------------------------------------------- per-class tiled WMMA NMS ----
__global__ __launch_bounds__(256)
void nms_kernel(const unsigned char* __restrict__ maskM,
                const int* __restrict__ orderAll,
                const float* __restrict__ scoreSorted,
                float* __restrict__ flatScore,   // [C, NN]
                int* __restrict__ flatGi) {      // [C, NN]
  __shared__ int order_s[NN];
  __shared__ unsigned char sup_s[NN];
  __shared__ unsigned int rowsD[16 * RS_D];      // 16 staged mask rows
  __shared__ int kept_s[16];
  __shared__ int valid_s[16];
  __shared__ unsigned int keptPack_s[4];
  __shared__ int keptCount;

  unsigned char* rowsB = (unsigned char*)rowsD;

  int c = blockIdx.x;
  int t = threadIdx.x;
  int lane = t & 31;
  int wave = t >> 5;

  for (int p = t; p < NN; p += 256) {
    int g = orderAll[c * NN + p];
    order_s[p] = g;
    sup_s[p] = 0;
    flatGi[c * NN + p] = g;
  }
  if (t == 0) keptCount = 0;
  __syncthreads();

  // A matrix: all-ones bytes across all K=64 -> D[m][n] counts every B byte,
  // independent of exact K placement inside B's VGPR layout.
  v8i aOnes;
#pragma unroll
  for (int q = 0; q < 8; ++q) aOnes[q] = 0x01010101;

  for (int tile = 0; tile < NTILES; ++tile) {
    int base = tile * 16;

    // ---- stage this tile's 16 mask rows into LDS (async global->LDS) ----
#pragma unroll
    for (int k = 0; k < 16; ++k) {
      const unsigned int* src =
          (const unsigned int*)(maskM + (size_t)order_s[base + k] * NN);
      for (int d = t; d < NN / 4; d += 256) {
#if USE_ASYNC_LDS
        unsigned int ldsOff =
            (unsigned int)(unsigned long long)&rowsD[k * RS_D + d];
        unsigned long long ga = (unsigned long long)(const void*)(src + d);
        asm volatile("global_load_async_to_lds_b32 %0, %1, off"
                     :: "v"(ldsOff), "v"(ga) : "memory");
#else
        rowsD[k * RS_D + d] = src[d];
#endif
      }
    }
    // prefetch next tile's rows toward cache (global_prefetch_b8)
    if (t < 16 && tile + 1 < NTILES) {
      int gnext = order_s[(tile + 1) * 16 + t];
      __builtin_prefetch((const void*)(maskM + (size_t)gnext * NN), 0, 1);
    }
#if USE_ASYNC_LDS
    asm volatile("s_wait_asynccnt 0x0" ::: "memory");
#endif
    __syncthreads();

    // ---- serial intra-tile NMS on wave 0 (16-step chain, LDS reads) ----
    if (t < 32) {
      int col = lane & 15;
      int pos = base + col;
      int gi = order_s[pos];
      int mySup = (int)sup_s[pos];
      int myKept = 0;
#pragma unroll
      for (int a = 0; a < 16; ++a) {
        int supA = __shfl(mySup, a, 32);
        int keptA = (supA == 0) ? 1 : 0;
        if (lane == a) myKept = keptA;
        int m = (int)rowsB[a * RS_B + gi];   // mask[gi_a][gi_lane]
        int later = (lane > a) ? 1 : 0;
        mySup |= (keptA & later & m);
      }
      if (lane < 16) { kept_s[lane] = myKept; }
    }
    __syncthreads();

    // ---- rank / valid (<= MAX_PER_CLASS) + kept byte-pack ----
    if (t == 0) {
      int cnt = keptCount;
      unsigned int pk0 = 0, pk1 = 0, pk2 = 0, pk3 = 0;
      for (int a = 0; a < 16; ++a) {
        if (kept_s[a]) {
          ++cnt;
          valid_s[a] = (cnt <= MAX_PER_CLASS) ? 1 : 0;
          unsigned int by = 0xFFu << ((a & 3) * 8);
          switch (a >> 2) {
            case 0: pk0 |= by; break;
            case 1: pk1 |= by; break;
            case 2: pk2 |= by; break;
            default: pk3 |= by; break;
          }
        } else {
          valid_s[a] = 0;
        }
      }
      keptCount = cnt;
      keptPack_s[0] = pk0; keptPack_s[1] = pk1;
      keptPack_s[2] = pk2; keptPack_s[3] = pk3;
    }
    __syncthreads();

    if (t < 16) {
      int pos = base + t;
      float s = scoreSorted[c * NN + pos];
      flatScore[c * NN + pos] = valid_s[t] ? s : NEGV;
    }

    unsigned int hiMask = (lane < 16) ? 0xFFFFFFFFu : 0u;
    unsigned int kp0 = keptPack_s[0] & hiMask;
    unsigned int kp1 = keptPack_s[1] & hiMask;
    unsigned int kp2 = keptPack_s[2] & hiMask;
    unsigned int kp3 = keptPack_s[3] & hiMask;

    // ---- WMMA suppression propagation: 2 column blocks per iteration ----
    // Pairing fills each WMMA's result-read hazard window with the other
    // block's pack / compare / store work instead of v_nops.
    int cbase = tile + 1;
    int nb = NTILES - cbase;                 // remaining column blocks
    int n = lane & 15;
    for (int q = wave * 2; q < nb; q += 16) {
      int q1 = (q + 1 < nb) ? (q + 1) : q;   // clamp odd tail (dup, no store)
      int cp0 = (cbase + q) * 16 + n;
      int cp1 = (cbase + q1) * 16 + n;
      int gj0 = order_s[cp0];
      int gj1 = order_s[cp1];
      unsigned int old0 = sup_s[cp0];
      unsigned int old1 = sup_s[cp1];
      unsigned int x0, x1, x2, x3, y0, y1, y2, y3;
      packCol(rowsB + gj0, x0, x1, x2, x3);
      packCol(rowsB + gj1, y0, y1, y2, y3);
      v8i b0 = { (int)(x0 & kp0), (int)(x1 & kp1),
                 (int)(x2 & kp2), (int)(x3 & kp3), 0, 0, 0, 0 };
      v8i b1 = { (int)(y0 & kp0), (int)(y1 & kp1),
                 (int)(y2 & kp2), (int)(y3 & kp3), 0, 0, 0, 0 };
      v8i cz = { 0, 0, 0, 0, 0, 0, 0, 0 };
      v8i d0 = __builtin_amdgcn_wmma_i32_16x16x64_iu8(false, aOnes, false, b0,
                                                      cz, false, false);
      v8i d1 = __builtin_amdgcn_wmma_i32_16x16x64_iu8(false, aOnes, false, b1,
                                                      cz, false, false);
      // lanes n and n+16 see identical counts (all D rows equal):
      // unconditional same-value RMW, branchless.
      sup_s[cp0] = (unsigned char)((old0 | (d0[0] != 0 ? 1u : 0u)) & 1u);
      if (q + 1 < nb) {                      // wave-uniform guard
        sup_s[cp1] = (unsigned char)((old1 | (d1[0] != 0 ? 1u : 0u)) & 1u);
      }
    }
    __syncthreads();
  }
}

// --------------------------------------------------- top-100 selection ----
__global__ __launch_bounds__(256)
void topk_kernel(float* __restrict__ flatScore,     // [C*NN], mutated
                 const int* __restrict__ flatGi,    // [C*NN]
                 const float4* __restrict__ boxes,  // [NN]
                 float* __restrict__ outBoxes,      // [100*4]
                 int* __restrict__ outCls) {        // [100]
  __shared__ float rs[256];
  __shared__ int ri[256];
  int t = threadIdx.x;
  const int TOT = NUM_CLASSES * NN;
  for (int r = 0; r < MAX_PER_IMAGE; ++r) {
    float bs = -FLT_MAX;
    int bi = TOT;
    for (int p = t; p < TOT; p += 256) {
      float s = flatScore[p];
      if (s > bs || (s == bs && p < bi)) { bs = s; bi = p; }
    }
    rs[t] = bs; ri[t] = bi;
    __syncthreads();
    for (int off = 128; off > 0; off >>= 1) {
      if (t < off) {
        float so = rs[t + off]; int io = ri[t + off];
        if (so > rs[t] || (so == rs[t] && io < ri[t])) { rs[t] = so; ri[t] = io; }
      }
      __syncthreads();
    }
    if (t == 0) {
      int fi = ri[0];
      int cls = fi / NN;
      int g = flatGi[fi];
      float4 bx = boxes[g];
      outBoxes[r * 4 + 0] = bx.x;
      outBoxes[r * 4 + 1] = bx.y;
      outBoxes[r * 4 + 2] = bx.z;
      outBoxes[r * 4 + 3] = bx.w;
      outCls[r] = cls;
      flatScore[fi] = -FLT_MAX;  // rewritten by nms_kernel each launch
    }
    __threadfence_block();
    __syncthreads();
  }
}

// ------------------------------------------------------------- launcher ----
extern "C" void kernel_launch(void* const* d_in, const int* in_sizes, int n_in,
                              void* d_out, int out_size, void* d_ws,
                              size_t ws_size, hipStream_t stream) {
  const float* props  = (const float*)d_in[0];  // [2000,4]
  const float* deltas = (const float*)d_in[1];  // [2000,4]
  const float* score  = (const float*)d_in[2];  // [2000,21]

  char* w = (char*)d_ws;
  float4* boxes        = (float4*)(w + 0);              //  32 KB
  unsigned char* maskM = (unsigned char*)(w + 32768);   //   4 MB
  int* orderAll        = (int*)(w + 4032768);           // 168 KB
  float* scoreSorted   = (float*)(w + 4200768);         // 168 KB
  float* flatScore     = (float*)(w + 4368768);         // 168 KB
  int* flatGi          = (int*)(w + 4536768);           // 168 KB

  float* outBoxes = (float*)d_out;          // first 400 floats
  int* outCls     = ((int*)d_out) + 400;    // next 100 int32

  decode_kernel<<<(NN + 255) / 256, 256, 0, stream>>>(props, deltas, boxes);

  long long tot = (long long)NN * NN;
  int iouBlocks = (int)((tot + 255) / 256);
  iou_mask_kernel<<<iouBlocks, 256, 0, stream>>>(boxes, maskM);

  sort_kernel<<<NUM_CLASSES, 1024, 0, stream>>>(score, orderAll, scoreSorted);

  nms_kernel<<<NUM_CLASSES, 256, 0, stream>>>(maskM, orderAll, scoreSorted,
                                              flatScore, flatGi);

  topk_kernel<<<1, 256, 0, stream>>>(flatScore, flatGi, boxes, outBoxes,
                                     outCls);
}
